// LR_GCN_79568564126322
// MI455X (gfx1250) — matrix operand order repaired and checked by reading, not compiled, not run on hardware
//
#include <hip/hip_runtime.h>

typedef __attribute__((ext_vector_type(16))) __bf16 v16bf;
typedef __attribute__((ext_vector_type(8)))  float  v8f;
typedef __attribute__((ext_vector_type(4)))  float  f4;
typedef __attribute__((ext_vector_type(4)))  int    i4;

#define WAVES_PER_BLOCK 8

__device__ __forceinline__ f4 ld4(const float* p) { return *(const f4*)p; }

__device__ __forceinline__ void pack4(v16bf& a, int base, f4 q) {
#pragma unroll
  for (int i = 0; i < 4; ++i) a[base + i] = (__bf16)q[i];
}

// B-operand fragment for v_wmma_f32_16x16x32_bf16: lane owns column n=lane&15;
// lanes 0-15 hold K=chunk*32+{0..7,16..23}, lanes 16-31 hold K=chunk*32+{8..15,24..31}
// (VGPR v<4 -> K pair 8h+2v; v>=4 -> 16+8h+2(v-4)). Branchless: clamped address +
// value select instead of exec-masked loads.
template<int KCAT>
__device__ __forceinline__ v16bf load_w_frag(const float* __restrict__ w,
                                             int chunk, int n, int lane) {
  v16bf b;
  const int kh = (lane >= 16) ? 8 : 0;
#pragma unroll
  for (int v = 0; v < 8; ++v) {
    const int k0 = chunk * 32 + kh + ((v & 4) << 2) + 2 * (v & 3);
    const int c0 = (k0     < KCAT) ? k0     : 0;
    const int c1 = (k0 + 1 < KCAT) ? k0 + 1 : 0;
    float f0 = w[c0 * 16 + n];
    float f1 = w[c1 * 16 + n];
    f0 = (k0     < KCAT) ? f0 : 0.0f;
    f1 = (k0 + 1 < KCAT) ? f1 : 0.0f;
    b[2 * v]     = (__bf16)f0;
    b[2 * v + 1] = (__bf16)f1;
  }
  return b;
}

// Shared tail: bias+ReLU on layer-1 accumulator, C-layout -> A-layout transpose
// through per-wave LDS (same-wave DS ops are processed in order; no barrier),
// second 16x16 WMMA, then segment-max scatter. Values are >= 0 and `out` is
// zero-initialized, so u32 max over the IEEE bit pattern == float max (and it
// reproduces the reference's 0-clamp of empty segments exactly).
__device__ __forceinline__ void mlp2_and_scatter(
    v8f acc, float bias1, float bias2, v16bf w2f,
    __bf16* __restrict__ ldsh, const int* __restrict__ ldssrc,
    int n, int khA, int mh, long long e0, long long E,
    float* __restrict__ out) {
#pragma unroll
  for (int r = 0; r < 8; ++r) {
    const float t = fmaxf(acc[r] + bias1, 0.0f);
    ldsh[(r + mh) * 16 + n] = (__bf16)t;
  }
  v16bf a2 = {};
#pragma unroll
  for (int v = 0; v < 4; ++v) {          // K=16 real; upper K-half stays zero
    const int k0 = khA + (v << 1);
    a2[2 * v]     = ldsh[n * 16 + k0];
    a2[2 * v + 1] = ldsh[n * 16 + k0 + 1];
  }
  v8f acc2 = {};
  acc2 = __builtin_amdgcn_wmma_f32_16x16x32_bf16(false, a2, false, w2f,
                                                 (short)0, acc2, false, false);

  // Two b128 LDS loads fetch all 8 src indices this lane scatters to.
  const i4 s0 = *(const i4*)(ldssrc + mh);
  const i4 s1 = *(const i4*)(ldssrc + mh + 4);

  if (e0 + 16 <= E) {                    // full tile: straight-line atomics
#pragma unroll
    for (int r = 0; r < 8; ++r) {
      const int   s = (r < 4) ? s0[r & 3] : s1[r & 3];
      const float t = fmaxf(acc2[r] + bias2, 0.0f);
      atomicMax((unsigned int*)(out + (size_t)s * 16 + n), __float_as_uint(t));
    }
  } else {                               // ragged tail tile
#pragma unroll
    for (int r = 0; r < 8; ++r) {
      if (e0 + (r + mh) < E) {
        const int   s = (r < 4) ? s0[r & 3] : s1[r & 3];
        const float t = fmaxf(acc2[r] + bias2, 0.0f);
        atomicMax((unsigned int*)(out + (size_t)s * 16 + n), __float_as_uint(t));
      }
    }
  }
}

// conv1: IN=4, KCAT=16. A chunk0: lanes 0-15 hold K=0..7 = x_src|x_dst,
// lanes 16-31 hold K=8..15 = edge_attr -> two b128 loads per lane.
__global__ __launch_bounds__(256) void lrconv1_wmma(
    const float* __restrict__ x,      // [N,4]
    const int*   __restrict__ ei,     // [2,E]
    const float* __restrict__ ea,     // [E,8]
    const float* __restrict__ w1, const float* __restrict__ b1,
    const float* __restrict__ w2, const float* __restrict__ b2,
    float* __restrict__ out,          // [N,16] pre-zeroed
    long long E) {
  __shared__ __bf16 lds_h[WAVES_PER_BLOCK][256];
  __shared__ __align__(16) int lds_src[WAVES_PER_BLOCK][16];

  const int  lane   = threadIdx.x & 31;
  const int  wave   = threadIdx.x >> 5;
  const int  n      = lane & 15;
  const bool lanelo = lane < 16;
  const int  khA    = lanelo ? 0 : 8;
  const int  mh     = lanelo ? 0 : 8;

  const v16bf w1f = load_w_frag<16>(w1, 0, n, lane);
  const v16bf w2f = load_w_frag<16>(w2, 0, n, lane);
  const float bias1 = b1[n];
  const float bias2 = b2[n];

  const long long nTiles  = (E + 15) >> 4;
  const long long tstride = (long long)gridDim.x * WAVES_PER_BLOCK;
  for (long long tile = (long long)blockIdx.x * WAVES_PER_BLOCK + wave;
       tile < nTiles; tile += tstride) {
    const long long e0 = tile << 4;
    long long e = e0 + n; if (e >= E) e = E - 1;
    const int src = ei[e];
    const int dst = ei[E + e];
    lds_src[wave][n] = src;             // both lane halves write identical data
    __builtin_prefetch(ei + e0 + (tstride << 4), 0, 1);

    const float* pA = lanelo ? x + (size_t)src * 4 : ea + e * 8;
    const float* pB = lanelo ? x + (size_t)dst * 4 : ea + e * 8 + 4;
    v16bf a = {};                        // VGPR4-7 (K=16..31) stay zero
    pack4(a, 0, ld4(pA));
    pack4(a, 4, ld4(pB));

    v8f acc = {};
    acc = __builtin_amdgcn_wmma_f32_16x16x32_bf16(false, a, false, w1f,
                                                  (short)0, acc, false, false);
    mlp2_and_scatter(acc, bias1, bias2, w2f, lds_h[wave], lds_src[wave],
                     n, khA, mh, e0, E, out);
  }
}

// conv2: IN=16, KCAT=40 -> two K=32 chunks.
// chunk0: VGPR0-3 = h_src[khA..khA+7], VGPR4-7 = h_dst[khA..khA+7].
// chunk1: lanes 0-15 carry edge_attr in K=32..39; everything else zero.
__global__ __launch_bounds__(256) void lrconv2_wmma(
    const float* __restrict__ h,      // [N,16]
    const int*   __restrict__ ei,
    const float* __restrict__ ea,
    const float* __restrict__ w1, const float* __restrict__ b1,  // [40,16]
    const float* __restrict__ w2, const float* __restrict__ b2,  // [16,16]
    float* __restrict__ out, long long E) {
  __shared__ __bf16 lds_h[WAVES_PER_BLOCK][256];
  __shared__ __align__(16) int lds_src[WAVES_PER_BLOCK][16];

  const int  lane   = threadIdx.x & 31;
  const int  wave   = threadIdx.x >> 5;
  const int  n      = lane & 15;
  const bool lanelo = lane < 16;
  const int  khA    = lanelo ? 0 : 8;
  const int  mh     = lanelo ? 0 : 8;

  const v16bf w1f0 = load_w_frag<40>(w1, 0, n, lane);
  const v16bf w1f1 = load_w_frag<40>(w1, 1, n, lane);
  const v16bf w2f  = load_w_frag<16>(w2, 0, n, lane);
  const float bias1 = b1[n];
  const float bias2 = b2[n];

  const long long nTiles  = (E + 15) >> 4;
  const long long tstride = (long long)gridDim.x * WAVES_PER_BLOCK;
  for (long long tile = (long long)blockIdx.x * WAVES_PER_BLOCK + wave;
       tile < nTiles; tile += tstride) {
    const long long e0 = tile << 4;
    long long e = e0 + n; if (e >= E) e = E - 1;
    const int src = ei[e];
    const int dst = ei[E + e];
    lds_src[wave][n] = src;
    __builtin_prefetch(ei + e0 + (tstride << 4), 0, 1);

    const float* ps = h + (size_t)src * 16 + khA;
    const float* pd = h + (size_t)dst * 16 + khA;
    v16bf a0 = {};
    pack4(a0, 0,  ld4(ps));
    pack4(a0, 4,  ld4(ps + 4));
    pack4(a0, 8,  ld4(pd));
    pack4(a0, 12, ld4(pd + 4));

    const f4 q0 = ld4(ea + e * 8);
    const f4 q1 = ld4(ea + e * 8 + 4);
    v16bf a1 = {};                       // K=40..63 stay zero
#pragma unroll
    for (int i = 0; i < 4; ++i) {
      a1[i]     = (__bf16)(lanelo ? q0[i] : 0.0f);
      a1[4 + i] = (__bf16)(lanelo ? q1[i] : 0.0f);
    }

    v8f acc = {};
    acc = __builtin_amdgcn_wmma_f32_16x16x32_bf16(false, a0, false, w1f0,
                                                  (short)0, acc, false, false);
    acc = __builtin_amdgcn_wmma_f32_16x16x32_bf16(false, a1, false, w1f1,
                                                  (short)0, acc, false, false);
    mlp2_and_scatter(acc, bias1, bias2, w2f, lds_h[wave], lds_src[wave],
                     n, khA, mh, e0, E, out);
  }
}

__global__ void zero_f32(float* __restrict__ p, long long nElem) {
  long long i = (long long)blockIdx.x * blockDim.x + threadIdx.x;
  const long long stride = (long long)gridDim.x * blockDim.x;
  for (; i < nElem; i += stride) p[i] = 0.0f;
}

// Node head: out = relu(h @ l1_w + l1_b) @ l2_w + l2_b  -> [N,1]
__global__ __launch_bounds__(256) void head_kernel(
    const float* __restrict__ h,
    const float* __restrict__ l1w, const float* __restrict__ l1b,
    const float* __restrict__ l2w, const float* __restrict__ l2b,
    float* __restrict__ out, int N) {
  const int i = blockIdx.x * blockDim.x + threadIdx.x;
  if (i >= N) return;
  float hv[16];
#pragma unroll
  for (int k = 0; k < 16; ++k) hv[k] = h[(size_t)i * 16 + k];
  float o = l2b[0];
#pragma unroll
  for (int j = 0; j < 16; ++j) {
    float t = l1b[j];
#pragma unroll
    for (int k = 0; k < 16; ++k) t = fmaf(hv[k], l1w[k * 16 + j], t);
    o = fmaf(fmaxf(t, 0.0f), l2w[j], o);
  }
  out[i] = o;
}

extern "C" void kernel_launch(void* const* d_in, const int* in_sizes, int n_in,
                              void* d_out, int out_size, void* d_ws, size_t ws_size,
                              hipStream_t stream) {
  const float* x    = (const float*)d_in[0];
  const int*   ei   = (const int*)  d_in[1];
  const float* ea   = (const float*)d_in[2];
  const float* c1w1 = (const float*)d_in[3];
  const float* c1b1 = (const float*)d_in[4];
  const float* c1w2 = (const float*)d_in[5];
  const float* c1b2 = (const float*)d_in[6];
  const float* c2w1 = (const float*)d_in[7];
  const float* c2b1 = (const float*)d_in[8];
  const float* c2w2 = (const float*)d_in[9];
  const float* c2b2 = (const float*)d_in[10];
  const float* l1w  = (const float*)d_in[11];
  const float* l1b  = (const float*)d_in[12];
  const float* l2w  = (const float*)d_in[13];
  const float* l2b  = (const float*)d_in[14];
  float* out = (float*)d_out;

  const int       N = in_sizes[0] / 4;
  const long long E = (long long)in_sizes[2] / 8;

  float* h1 = (float*)d_ws;                 // [N,16]
  float* h2 = h1 + (size_t)N * 16;          // [N,16]
  const long long hElems = (long long)N * 16;

  zero_f32<<<512, 256, 0, stream>>>(h1, hElems);
  lrconv1_wmma<<<2048, 256, 0, stream>>>(x, ei, ea, c1w1, c1b1, c1w2, c1b2, h1, E);
  zero_f32<<<512, 256, 0, stream>>>(h2, hElems);
  lrconv2_wmma<<<2048, 256, 0, stream>>>(h1, ei, ea, c2w1, c2b1, c2w2, c2b2, h2, E);
  head_kernel<<<(N + 255) / 256, 256, 0, stream>>>(h2, l1w, l1b, l2w, l2b, out, N);
}
